// Res16UNetBackbone_44040594653570
// MI455X (gfx1250) — compile-verified, matrix operand order checked by reading, not run on hardware
//
#include <hip/hip_runtime.h>

typedef __attribute__((ext_vector_type(16))) _Float16 v16h;
typedef __attribute__((ext_vector_type(8)))  float    v8f;

#define TPB 256
#define CONV_TPB 128
#define CONV_WAVES 4
#define MAX_TPW 4   // up to Cout=256 -> 16 n-tiles / 4 waves

// ---------------------------------------------------------------------------
// Device kernels
// ---------------------------------------------------------------------------

__global__ void kzero_i(int* __restrict__ p, long n) {
  long i = (long)blockIdx.x * blockDim.x + threadIdx.x;
  if (i < n) p[i] = 0;
}

// f32 [taps,Cin,Cout] -> f16 WMMA B-fragment layout:
// dst[((tap*nkk + kk)*ntiles + jj)*512 + lane*16 + h]
//   = W[tap][ kk*32 + kb + (h<8 ? h : h+8) ][ jj*16 + (lane&15) ]   (kb=(lane>>4)*8)
// zero-padded for K >= Cin.  Each B fragment is then one 32B/lane vector load.
__global__ void kcvtw_frag(const float* __restrict__ src, _Float16* __restrict__ dst,
                           int taps, int Cin, int Cinp, int Cout) {
  int nkk = Cinp >> 5;
  int ntiles = Cout >> 4;
  long total = (long)taps * Cinp * Cout;   // == taps*nkk*ntiles*512
  long t = (long)blockIdx.x * blockDim.x + threadIdx.x;
  if (t >= total) return;
  int h = (int)(t & 15);
  int lane = (int)((t >> 4) & 31);
  long frag = t >> 9;
  int jj = (int)(frag % ntiles);
  long f2 = frag / ntiles;
  int kk = (int)(f2 % nkk);
  int tap = (int)(f2 / nkk);
  int kb = (lane >> 4) << 3;
  int n = (jj << 4) + (lane & 15);
  int Krel = (kk << 5) + kb + ((h < 8) ? h : h + 8);
  dst[t] = (Krel < Cin) ? (_Float16)src[((long)tap * Cin + Krel) * Cout + n]
                        : (_Float16)0.f;
}

// Generic gather-GEMM conv:  out[N,Cout] = sum_k gather(x, nbr[:,k]) @ W[k]
// W is f16 in fragment layout (see kcvtw_frag). nbr == nullptr => identity gather.
// Block: 128 threads = 4 waves; block owns a 32-row tile (2 M-subtiles).
// A is staged in LDS directly in WMMA fragment layout so each A fragment is one
// contiguous 32B/lane LDS read; each B fragment (one 32B/lane global read) feeds
// two WMMAs.
__global__ __launch_bounds__(CONV_TPB) void kconv_wmma(
    const float* __restrict__ x, const int* __restrict__ nbr, int taps,
    const _Float16* __restrict__ W, float* __restrict__ out,
    int N, int Cin, int Cinp, int Cout) {
  __shared__ _Float16 As[2 * 8 * 512];   // 16KB: [(mt*nkk + kk)][lane][h]
  const int row0 = blockIdx.x * 32;
  const int tid  = threadIdx.x;
  const int lane = tid & 31;
  const int wave = tid >> 5;
  const int ntiles = Cout >> 4;
  const int nkk    = Cinp >> 5;

  // staging: 4 threads per row, 32 rows
  const int sr  = tid >> 2;       // row 0..31
  const int sc  = tid & 3;        // column start
  const int srow = row0 + sr;
  const int smt = sr >> 4;        // M subtile
  const int srr = sr & 15;        // row within subtile

  v8f acc[2][MAX_TPW] = {};

  for (int k = 0; k < taps; ++k) {
    __syncthreads();  // previous-iter readers done before restaging
    // stage gathered row (f32 -> f16) into fragment-swizzled LDS
    int idx = -1;
    if (srow < N) idx = nbr ? nbr[(long)srow * taps + k] : srow;
    const float* xr = x + (long)(idx < 0 ? 0 : idx) * Cin;
    for (int c = sc; c < Cinp; c += 4) {
      float v = (idx >= 0 && c < Cin) ? xr[c] : 0.f;
      int cc = c & 31;
      int l = srr + (((cc >> 3) & 1) << 4);          // owning lane
      int h = (cc & 7) + ((cc >> 4) << 3);           // half index within lane
      As[((smt * nkk + (c >> 5)) << 9) + (l << 4) + h] = (_Float16)v;
    }
    __syncthreads();

    for (int kk = 0; kk < nkk; ++kk) {
      // A fragments: one contiguous 32B LDS read per M subtile
      v16h a0 = *(const v16h*)&As[(kk << 9) + (lane << 4)];
      v16h a1 = *(const v16h*)&As[((nkk + kk) << 9) + (lane << 4)];
#pragma unroll
      for (int t = 0; t < MAX_TPW; ++t) {
        int jj = wave + t * CONV_WAVES;   // wave-uniform guard, EXEC stays full
        if (jj < ntiles) {
          // B fragment: one aligned 32B vector load per lane (pre-swizzled)
          v16h b = *(const v16h*)(
              W + (((long)k * nkk + kk) * ntiles + jj) * 512 + (lane << 4));
          acc[0][t] = __builtin_amdgcn_wmma_f32_16x16x32_f16(
              false, a0, false, b, (short)0, acc[0][t], false, false);
          acc[1][t] = __builtin_amdgcn_wmma_f32_16x16x32_f16(
              false, a1, false, b, (short)0, acc[1][t], false, false);
        }
      }
    }
  }

  // D layout: lanes 0-15 -> M=i, N=lane; lanes 16-31 -> M=8+i, N=lane-16
  const int mb = (lane >> 4) << 3;
  const int nc = lane & 15;
#pragma unroll
  for (int mt = 0; mt < 2; ++mt) {
#pragma unroll
    for (int t = 0; t < MAX_TPW; ++t) {
      int jj = wave + t * CONV_WAVES;
      if (jj < ntiles) {
        int col = (jj << 4) + nc;
#pragma unroll
        for (int i = 0; i < 8; ++i) {
          int row = row0 + mt * 16 + mb + i;
          if (row < N) out[(long)row * Cout + col] = acc[mt][t][i];
        }
      }
    }
  }
}

// Two-pass per-channel batchnorm statistics (deterministic tree reduction)
__global__ void kbn_stats(const float* __restrict__ x, int N, int C,
                          float* __restrict__ mean, float* __restrict__ rstd) {
  int c = blockIdx.x;
  __shared__ float red[TPB];
  __shared__ float mS;
  float s = 0.f;
  for (int i = threadIdx.x; i < N; i += TPB) s += x[(long)i * C + c];
  red[threadIdx.x] = s;
  __syncthreads();
  for (int w = TPB / 2; w > 0; w >>= 1) {
    if (threadIdx.x < w) red[threadIdx.x] += red[threadIdx.x + w];
    __syncthreads();
  }
  if (threadIdx.x == 0) mS = red[0] / (float)N;
  __syncthreads();
  float m = mS;
  float v = 0.f;
  for (int i = threadIdx.x; i < N; i += TPB) {
    float d = x[(long)i * C + c] - m;
    v += d * d;
  }
  red[threadIdx.x] = v;
  __syncthreads();
  for (int w = TPB / 2; w > 0; w >>= 1) {
    if (threadIdx.x < w) red[threadIdx.x] += red[threadIdx.x + w];
    __syncthreads();
  }
  if (threadIdx.x == 0) {
    mean[c] = m;
    rstd[c] = rsqrtf(red[0] / (float)N + 1e-5f);
  }
}

// y = g*(x-m)*rstd + b [+res] [relu]
__global__ void kbn_apply(const float* __restrict__ x, const float* __restrict__ mean,
                          const float* __restrict__ rstd, const float* __restrict__ g,
                          const float* __restrict__ b, const float* __restrict__ res,
                          float* __restrict__ out, long total, int C, int relu) {
  long i = (long)blockIdx.x * blockDim.x + threadIdx.x;
  if (i >= total) return;
  int c = (int)(i % C);
  float v = g[c] * (x[i] - mean[c]) * rstd[c] + b[c];
  if (res) v += res[i];
  if (relu) v = fmaxf(v, 0.f);
  out[i] = v;
}

// Build per-parent child lists (set is order-independent; summed sorted later)
__global__ void kdown_build(const int* __restrict__ parent, int Nf,
                            int* __restrict__ counts, int* __restrict__ child) {
  int i = blockIdx.x * blockDim.x + threadIdx.x;
  if (i >= Nf) return;
  int p = parent[i];
  int pos = atomicAdd(&counts[p], 1);
  if (pos < 8) child[p * 8 + pos] = i;
}

// out[p,co] = sum over sorted children i of parent p:  x[i] @ W[koff[i]]
__global__ void kdown_apply(const float* __restrict__ x, const int* __restrict__ koff,
                            const int* __restrict__ counts, const int* __restrict__ child,
                            const float* __restrict__ W, float* __restrict__ out,
                            int Nc, int Cin, int Cout) {
  long t = (long)blockIdx.x * blockDim.x + threadIdx.x;
  long tot = (long)Nc * Cout;
  if (t >= tot) return;
  int p  = (int)(t / Cout);
  int co = (int)(t % Cout);
  int cnt = counts[p];
  if (cnt > 8) cnt = 8;
  int ch[8];
  for (int j = 0; j < cnt; ++j) ch[j] = child[p * 8 + j];
  for (int a = 1; a < cnt; ++a) {           // insertion sort -> deterministic order
    int v = ch[a]; int b2 = a - 1;
    while (b2 >= 0 && ch[b2] > v) { ch[b2 + 1] = ch[b2]; --b2; }
    ch[b2 + 1] = v;
  }
  float acc = 0.f;
  for (int j = 0; j < cnt; ++j) {
    int i = ch[j];
    const float* wr = W + (long)koff[i] * Cin * Cout + co;
    const float* xr = x + (long)i * Cin;
    for (int ci = 0; ci < Cin; ++ci) acc += xr[ci] * wr[(long)ci * Cout];
  }
  out[t] = acc;
}

// out[i,co] = xc[parent[i]] @ W[koff[i]]
__global__ void kup_apply(const float* __restrict__ xc, const int* __restrict__ parent,
                          const int* __restrict__ koff, const float* __restrict__ W,
                          float* __restrict__ out, int Nf, int Cin, int Cout) {
  long t = (long)blockIdx.x * blockDim.x + threadIdx.x;
  long tot = (long)Nf * Cout;
  if (t >= tot) return;
  int i  = (int)(t / Cout);
  int co = (int)(t % Cout);
  const float* wr = W + (long)koff[i] * Cin * Cout + co;
  const float* xr = xc + (long)parent[i] * Cin;
  float acc = 0.f;
  for (int ci = 0; ci < Cin; ++ci) acc += xr[ci] * wr[(long)ci * Cout];
  out[t] = acc;
}

__global__ void kconcat(const float* __restrict__ a, int Ca, const float* __restrict__ b,
                        int Cb, float* __restrict__ out, long N) {
  long i = (long)blockIdx.x * blockDim.x + threadIdx.x;
  int C = Ca + Cb;
  long tot = N * C;
  if (i >= tot) return;
  long row = i / C;
  int c = (int)(i % C);
  out[i] = (c < Ca) ? a[row * Ca + c] : b[row * Cb + (c - Ca)];
}

// ---------------------------------------------------------------------------
// Host orchestration
// ---------------------------------------------------------------------------

namespace {

struct Arena {
  char* base;
  size_t off;
  void* raw(size_t bytes) {
    off = (off + 255) & ~(size_t)255;
    void* p = base + off;
    off += bytes;
    return p;
  }
  float* f(size_t n) { return (float*)raw(n * sizeof(float)); }
  int* i(size_t n) { return (int*)raw(n * sizeof(int)); }
  _Float16* h(size_t n) { return (_Float16*)raw(n * sizeof(_Float16)); }
};

static inline unsigned divup(long a, long b) { return (unsigned)((a + b - 1) / b); }
static inline int pad32(int c) { return (c + 31) & ~31; }

static void L_zero_i(hipStream_t s, int* p, long n) {
  kzero_i<<<divup(n, TPB), TPB, 0, s>>>(p, n);
}
static void L_cvtw(hipStream_t s, const float* src, _Float16* dst, int taps, int Cin,
                   int Cinp, int Cout) {
  long tot = (long)taps * Cinp * Cout;
  kcvtw_frag<<<divup(tot, TPB), TPB, 0, s>>>(src, dst, taps, Cin, Cinp, Cout);
}
static void L_conv(hipStream_t s, const float* x, const int* nbr, int taps,
                   const _Float16* W, float* out, int N, int Cin, int Cinp, int Cout) {
  kconv_wmma<<<divup(N, 32), CONV_TPB, 0, s>>>(x, nbr, taps, W, out, N, Cin, Cinp, Cout);
}
static void L_stats(hipStream_t s, const float* x, int N, int C, float* m, float* r) {
  kbn_stats<<<C, TPB, 0, s>>>(x, N, C, m, r);
}
static void L_apply(hipStream_t s, const float* x, const float* m, const float* r,
                    const float* g, const float* b, const float* res, float* out, long N,
                    int C, int relu) {
  long tot = N * C;
  kbn_apply<<<divup(tot, TPB), TPB, 0, s>>>(x, m, r, g, b, res, out, tot, C, relu);
}
static void L_concat(hipStream_t s, const float* a, int Ca, const float* b, int Cb,
                     float* out, long N) {
  long tot = N * (long)(Ca + Cb);
  kconcat<<<divup(tot, TPB), TPB, 0, s>>>(a, Ca, b, Cb, out, N);
}

struct BSpec { int base; bool ds; int Cin, Co; };
struct BP { const float *b1, *b2, *g1, *g2, *w1, *w2, *dsw, *dsg, *dsb; };

static BP make_bp(void* const* d, int base, bool ds) {
  BP p{};
  if (ds) {
    p.b1  = (const float*)d[base + 0];
    p.b2  = (const float*)d[base + 1];
    p.dsb = (const float*)d[base + 2];
    p.dsg = (const float*)d[base + 3];
    p.dsw = (const float*)d[base + 4];
    p.g1  = (const float*)d[base + 5];
    p.g2  = (const float*)d[base + 6];
    p.w1  = (const float*)d[base + 7];
    p.w2  = (const float*)d[base + 8];
  } else {
    p.b1 = (const float*)d[base + 0];
    p.b2 = (const float*)d[base + 1];
    p.g1 = (const float*)d[base + 2];
    p.g2 = (const float*)d[base + 3];
    p.w1 = (const float*)d[base + 4];
    p.w2 = (const float*)d[base + 5];
  }
  return p;
}

// Residual basic block: relu(bn2(conv2(relu(bn1(conv1(x))))) + idn)
static void run_block(Arena& ws, hipStream_t s, const float* x, int N, const int* nb,
                      const BSpec& sp, const BP& p, const _Float16* w1h,
                      const _Float16* w2h, const _Float16* dsh, float* out) {
  size_t mk = ws.off;
  int Cin = sp.Cin, Co = sp.Co;
  float* A  = ws.f((size_t)N * Co);
  float* B  = ws.f((size_t)N * Co);
  float* m1 = ws.f(Co);
  float* r1 = ws.f(Co);
  L_conv(s, x, nb, 27, w1h, A, N, Cin, pad32(Cin), Co);
  L_stats(s, A, N, Co, m1, r1);
  L_apply(s, A, m1, r1, p.g1, p.b1, nullptr, B, N, Co, 1);
  L_conv(s, B, nb, 27, w2h, A, N, Co, Co, Co);
  L_stats(s, A, N, Co, m1, r1);
  const float* resid = x;
  if (dsh) {
    float* Cb = ws.f((size_t)N * Co);
    float* m2 = ws.f(Co);
    float* r2 = ws.f(Co);
    L_conv(s, x, nullptr, 1, dsh, Cb, N, Cin, pad32(Cin), Co);
    L_stats(s, Cb, N, Co, m2, r2);
    L_apply(s, Cb, m2, r2, p.dsg, p.dsb, nullptr, Cb, N, Co, 0);
    resid = Cb;
  }
  L_apply(s, A, m1, r1, p.g2, p.b2, resid, out, N, Co, 1);
  ws.off = mk;
}

// Stride-2 kernel-2 down conv + BN + ReLU
static void run_down(Arena& ws, hipStream_t s, const float* x, int Nf, int Nc,
                     const int* parent, const int* koff, const float* W, const float* g,
                     const float* b, int Cin, int Co, float* out) {
  size_t mk = ws.off;
  int* counts = ws.i(Nc);
  int* child  = ws.i((size_t)Nc * 8);
  float* t = ws.f((size_t)Nc * Co);
  float* m = ws.f(Co);
  float* r = ws.f(Co);
  L_zero_i(s, counts, Nc);
  kdown_build<<<divup(Nf, TPB), TPB, 0, s>>>(parent, Nf, counts, child);
  long tot = (long)Nc * Co;
  kdown_apply<<<divup(tot, TPB), TPB, 0, s>>>(x, koff, counts, child, W, t, Nc, Cin, Co);
  L_stats(s, t, Nc, Co, m, r);
  L_apply(s, t, m, r, g, b, nullptr, out, Nc, Co, 1);
  ws.off = mk;
}

// Transposed stride-2 conv + BN + ReLU, then concat with skip
static void run_up_concat(Arena& ws, hipStream_t s, const float* xc, int Nf,
                          const int* parent, const int* koff, const float* W,
                          const float* g, const float* b, int Cin, int Co,
                          const float* skip, int Cskip, float* xcat) {
  size_t mk = ws.off;
  float* t = ws.f((size_t)Nf * Co);
  float* m = ws.f(Co);
  float* r = ws.f(Co);
  long tot = (long)Nf * Co;
  kup_apply<<<divup(tot, TPB), TPB, 0, s>>>(xc, parent, koff, W, t, Nf, Cin, Co);
  L_stats(s, t, Nf, Co, m, r);
  L_apply(s, t, m, r, g, b, nullptr, t, Nf, Co, 1);
  L_concat(s, t, Co, skip, Cskip, xcat, Nf);
  ws.off = mk;
}

}  // namespace

extern "C" void kernel_launch(void* const* d_in, const int* in_sizes, int n_in,
                              void* d_out, int out_size, void* d_ws, size_t ws_size,
                              hipStream_t stream) {
  (void)n_in; (void)out_size; (void)ws_size;
  hipStream_t s = stream;
  Arena ws{(char*)d_ws, 0};

  // --------- input pointer map (pytree-flattened params, see analysis) -------
  const float* feats = (const float*)d_in[0];
  const int* nbr[5];
  for (int l = 0; l < 5; ++l) nbr[l] = (const int*)d_in[146 + l];
  const int* parent[4];
  const int* koff[4];
  for (int l = 0; l < 4; ++l) {
    parent[l] = (const int*)d_in[151 + 2 * l];
    koff[l]   = (const int*)d_in[152 + 2 * l];
  }
  const int N0 = in_sizes[0] / 9;
  const int n1 = in_sizes[147] / 27;
  const int n2 = in_sizes[148] / 27;
  const int n3 = in_sizes[149] / 27;
  const int n4 = in_sizes[150] / 27;

  // bn0..bn4, bntr4..bntr7  (g,b pairs at 118..135)
  const float* bng[9];
  const float* bnb[9];
  for (int i = 0; i < 9; ++i) {
    bng[i] = (const float*)d_in[118 + 2 * i];
    bnb[i] = (const float*)d_in[119 + 2 * i];
  }
  const float* Wconv0 = (const float*)d_in[136];
  const float* Wdown[4] = {(const float*)d_in[137], (const float*)d_in[138],
                           (const float*)d_in[139], (const float*)d_in[140]};
  const float* Wfinal = (const float*)d_in[141];
  const float* Wup[4] = {(const float*)d_in[142], (const float*)d_in[143],
                         (const float*)d_in[144], (const float*)d_in[145]};

  const BSpec bs[16] = {
      {1, false, 32, 32},   {7, false, 32, 32},     // block1
      {13, true, 32, 64},   {22, false, 64, 64},    // block2
      {28, true, 64, 128},  {37, false, 128, 128},  // block3
      {43, true, 128, 256}, {52, false, 256, 256},  // block4
      {58, true, 256, 128}, {67, false, 128, 128},  // block5
      {73, true, 192, 128}, {82, false, 128, 128},  // block6
      {88, true, 128, 96},  {97, false, 96, 96},    // block7
      {103, true, 128, 96}, {112, false, 96, 96}};  // block8
  BP bp[16];
  for (int i = 0; i < 16; ++i) bp[i] = make_bp(d_in, bs[i].base, bs[i].ds);

  // --------- convert WMMA-path weights to fragment-layout f16 ----------------
  _Float16* hW1[16];
  _Float16* hW2[16];
  _Float16* hDS[16];
  for (int i = 0; i < 16; ++i) {
    const BSpec& sp = bs[i];
    hW1[i] = ws.h((size_t)27 * sp.Cin * sp.Co);
    L_cvtw(s, bp[i].w1, hW1[i], 27, sp.Cin, sp.Cin, sp.Co);
    hW2[i] = ws.h((size_t)27 * sp.Co * sp.Co);
    L_cvtw(s, bp[i].w2, hW2[i], 27, sp.Co, sp.Co, sp.Co);
    if (sp.ds) {
      hDS[i] = ws.h((size_t)sp.Cin * sp.Co);
      L_cvtw(s, bp[i].dsw, hDS[i], 1, sp.Cin, sp.Cin, sp.Co);
    } else {
      hDS[i] = nullptr;
    }
  }
  _Float16* hC0 = ws.h((size_t)27 * 32 * 32);   // conv0: Cin 9 padded to 32
  L_cvtw(s, Wconv0, hC0, 27, 9, 32, 32);
  _Float16* hFin = ws.h((size_t)96 * 256);
  L_cvtw(s, Wfinal, hFin, 1, 96, 96, 256);

  // --------- encoder ---------------------------------------------------------
  float* x0 = ws.f((size_t)N0 * 32);
  {
    size_t mk = ws.off;
    float* t = ws.f((size_t)N0 * 32);
    float* m = ws.f(32);
    float* r = ws.f(32);
    L_conv(s, feats, nbr[0], 27, hC0, t, N0, 9, 32, 32);
    L_stats(s, t, N0, 32, m, r);
    L_apply(s, t, m, r, bng[0], bnb[0], nullptr, x0, N0, 32, 1);
    ws.off = mk;
  }

  float* b1 = ws.f((size_t)n1 * 32);
  run_down(ws, s, x0, N0, n1, parent[0], koff[0], Wdown[0], bng[1], bnb[1], 32, 32, b1);
  run_block(ws, s, b1, n1, nbr[1], bs[0], bp[0], hW1[0], hW2[0], nullptr, b1);
  run_block(ws, s, b1, n1, nbr[1], bs[1], bp[1], hW1[1], hW2[1], nullptr, b1);

  float* b2 = ws.f((size_t)n2 * 64);
  {
    size_t mk = ws.off;
    float* t = ws.f((size_t)n2 * 32);
    run_down(ws, s, b1, n1, n2, parent[1], koff[1], Wdown[1], bng[2], bnb[2], 32, 32, t);
    run_block(ws, s, t, n2, nbr[2], bs[2], bp[2], hW1[2], hW2[2], hDS[2], b2);
    ws.off = mk;
  }
  run_block(ws, s, b2, n2, nbr[2], bs[3], bp[3], hW1[3], hW2[3], nullptr, b2);

  float* b3 = ws.f((size_t)n3 * 128);
  {
    size_t mk = ws.off;
    float* t = ws.f((size_t)n3 * 64);
    run_down(ws, s, b2, n2, n3, parent[2], koff[2], Wdown[2], bng[3], bnb[3], 64, 64, t);
    run_block(ws, s, t, n3, nbr[3], bs[4], bp[4], hW1[4], hW2[4], hDS[4], b3);
    ws.off = mk;
  }
  run_block(ws, s, b3, n3, nbr[3], bs[5], bp[5], hW1[5], hW2[5], nullptr, b3);

  float* b4 = ws.f((size_t)n4 * 256);
  {
    size_t mk = ws.off;
    float* t = ws.f((size_t)n4 * 128);
    run_down(ws, s, b3, n3, n4, parent[3], koff[3], Wdown[3], bng[4], bnb[4], 128, 128, t);
    run_block(ws, s, t, n4, nbr[4], bs[6], bp[6], hW1[6], hW2[6], hDS[6], b4);
    ws.off = mk;
  }
  run_block(ws, s, b4, n4, nbr[4], bs[7], bp[7], hW1[7], hW2[7], nullptr, b4);

  // --------- decoder ---------------------------------------------------------
  float* x3 = ws.f((size_t)n3 * 128);
  {
    size_t mk = ws.off;
    float* xcat = ws.f((size_t)n3 * 256);
    run_up_concat(ws, s, b4, n3, parent[3], koff[3], Wup[0], bng[5], bnb[5], 256, 128,
                  b3, 128, xcat);
    run_block(ws, s, xcat, n3, nbr[3], bs[8], bp[8], hW1[8], hW2[8], hDS[8], x3);
    ws.off = mk;
  }
  run_block(ws, s, x3, n3, nbr[3], bs[9], bp[9], hW1[9], hW2[9], nullptr, x3);

  float* x2 = ws.f((size_t)n2 * 128);
  {
    size_t mk = ws.off;
    float* xcat = ws.f((size_t)n2 * 192);
    run_up_concat(ws, s, x3, n2, parent[2], koff[2], Wup[1], bng[6], bnb[6], 128, 128,
                  b2, 64, xcat);
    run_block(ws, s, xcat, n2, nbr[2], bs[10], bp[10], hW1[10], hW2[10], hDS[10], x2);
    ws.off = mk;
  }
  run_block(ws, s, x2, n2, nbr[2], bs[11], bp[11], hW1[11], hW2[11], nullptr, x2);

  float* x1 = ws.f((size_t)n1 * 96);
  {
    size_t mk = ws.off;
    float* xcat = ws.f((size_t)n1 * 128);
    run_up_concat(ws, s, x2, n1, parent[1], koff[1], Wup[2], bng[7], bnb[7], 128, 96,
                  b1, 32, xcat);
    run_block(ws, s, xcat, n1, nbr[1], bs[12], bp[12], hW1[12], hW2[12], hDS[12], x1);
    ws.off = mk;
  }
  run_block(ws, s, x1, n1, nbr[1], bs[13], bp[13], hW1[13], hW2[13], nullptr, x1);

  float* xf = ws.f((size_t)N0 * 96);
  {
    size_t mk = ws.off;
    float* xcat = ws.f((size_t)N0 * 128);
    run_up_concat(ws, s, x1, N0, parent[0], koff[0], Wup[3], bng[8], bnb[8], 96, 96,
                  x0, 32, xcat);
    run_block(ws, s, xcat, N0, nbr[0], bs[14], bp[14], hW1[14], hW2[14], hDS[14], xf);
    ws.off = mk;
  }
  run_block(ws, s, xf, N0, nbr[0], bs[15], bp[15], hW1[15], hW2[15], nullptr, xf);

  // --------- final linear: [N0,96] @ [96,256] -> d_out -----------------------
  L_conv(s, xf, nullptr, 1, hFin, (float*)d_out, N0, 96, 96, 256);
}
